// ForegroundAttentionModule_50483045597785
// MI455X (gfx1250) — matrix operand order repaired and checked by reading, not compiled, not run on hardware
//
#include <hip/hip_runtime.h>
#include <math.h>

typedef __attribute__((ext_vector_type(16))) __bf16 v16bf;
typedef __attribute__((ext_vector_type(8)))  float  v8f;

#define DEV static __device__ __forceinline__

constexpr int B = 4, C = 512, T = 2048;
constexpr float SCALE  = 0.04419417382415922f; // 1/sqrt(512)
constexpr float LN_EPS = 1e-5f;

union FragBF { v16bf v; uint4 q[2]; };

// A-operand: row-major [M x K] tile, 16x32 bf16. Lane m holds K = kh*8..+7 and 16+kh*8..+7.
DEV v16bf load_a_frag(const __bf16* tile, int ld, int lane) {
    int m = lane & 15, kh = lane >> 4;
    const __bf16* r = tile + (size_t)m * ld + kh * 8;
    FragBF f;
    f.q[0] = *reinterpret_cast<const uint4*>(r);
    f.q[1] = *reinterpret_cast<const uint4*>(r + 16);
    return f.v;
}
// B-operand: we store B transposed (N-major row-major). Lane n holds K = kh*8..+7 and 16+kh*8..+7.
DEV v16bf load_b_frag(const __bf16* tileT, int ld, int lane) {
    int n = lane & 15, kh = lane >> 4;
    const __bf16* r = tileT + (size_t)n * ld + kh * 8;
    FragBF f;
    f.q[0] = *reinterpret_cast<const uint4*>(r);
    f.q[1] = *reinterpret_cast<const uint4*>(r + 16);
    return f.v;
}
DEV v8f wmma_bf16(v16bf a, v16bf b, v8f c) {
    return __builtin_amdgcn_wmma_f32_16x16x32_bf16(false, a, false, b, (short)0, c, false, false);
}

// ---------------------------------------------------------------- K0: weights f32 -> bf16
__global__ void convert_weights(const float* wq, const float* wk, const float* wv, const float* ws,
                                __bf16* oq, __bf16* ok, __bf16* ov, __bf16* os) {
    int i = blockIdx.x * blockDim.x + threadIdx.x;
    if (i < C * C) {
        oq[i] = (__bf16)wq[i]; ok[i] = (__bf16)wk[i];
        ov[i] = (__bf16)wv[i]; os[i] = (__bf16)ws[i];
    }
}

// ---------------------------------------------------------------- K1: QKV projections
// Qt[b][t][o], Kt[b][t][o] (bf16, row-major over o), Vc[b][o][t] (bf16, row-major over t).
// Also per-block partial of cb = sum_{t,o} value[t,o]*Wl[o] (f32 accumulators, pre-bf16-rounding).
__global__ __launch_bounds__(256) void qkv_kernel(
    const float* __restrict__ feat,
    const __bf16* __restrict__ Wq, const __bf16* __restrict__ Wk, const __bf16* __restrict__ Wv,
    const float* __restrict__ bq, const float* __restrict__ bk, const float* __restrict__ bv,
    const float* __restrict__ Wl,
    __bf16* __restrict__ Qt, __bf16* __restrict__ Kt, __bf16* __restrict__ Vc,
    float* __restrict__ vdot_part) {
    constexpr int TT = 32;       // t-tile per block
    constexpr int LDF = C + 8;   // 520 halves (row = 1040 B, 16B-aligned)
    __shared__ __bf16 featT[TT * LDF];
    __shared__ float  red[256];
    const int tid = threadIdx.x, lane = tid & 31, w = tid >> 5;
    const int b = blockIdx.y, t0 = blockIdx.x * TT;
    const float* fb = feat + (size_t)b * C * T;

    // stage feat^T tile [32 t x 512 c] as bf16 (coalesced global reads)
    for (int it = 0; it < (C * TT) / 256; ++it) {
        int idx = it * 256 + tid;
        int c = idx >> 5, tt = idx & 31;
        featT[tt * LDF + c] = (__bf16)fb[(size_t)c * T + t0 + tt];
    }
    __syncthreads();

    float vpart = 0.f;
    const int n = lane & 15, mh = (lane >> 4) << 3;
    for (int jb = w; jb < 3 * (C / 16); jb += 8) {     // 96 (matrix, n-tile) jobs
        const int mat = jb >> 5;                       // 0:Q 1:K 2:V
        const int n0 = (jb & 31) * 16;
        const __bf16* W   = (mat == 0) ? Wq : (mat == 1) ? Wk : Wv;
        const float* bias = (mat == 0) ? bq : (mat == 1) ? bk : bv;
        const float bvn = bias[n0 + n];
        const float wl  = (mat == 2) ? Wl[n0 + n] : 0.f;
        for (int msub = 0; msub < 2; ++msub) {
            v8f acc = {0, 0, 0, 0, 0, 0, 0, 0};
            for (int k = 0; k < C; k += 32) {
                v16bf a  = load_a_frag(featT + msub * 16 * LDF + k, LDF, lane);
                v16bf bb = load_b_frag(W + (size_t)n0 * C + k, C, lane);
                acc = wmma_bf16(a, bb, acc);
            }
            for (int i = 0; i < 8; ++i) {
                int t = t0 + msub * 16 + i + mh;
                float val = acc[i] + bvn;
                if (mat == 0)      Qt[((size_t)b * T + t) * C + n0 + n] = (__bf16)val;
                else if (mat == 1) Kt[((size_t)b * T + t) * C + n0 + n] = (__bf16)val;
                else { Vc[((size_t)b * C + n0 + n) * T + t] = (__bf16)val; vpart += val * wl; }
            }
        }
    }
    // deterministic block reduction of cb partial
    red[tid] = vpart; __syncthreads();
    for (int s = 128; s > 0; s >>= 1) { if (tid < s) red[tid] += red[tid + s]; __syncthreads(); }
    if (tid == 0) vdot_part[b * (T / TT) + blockIdx.x] = red[0];
}

// ---------------------------------------------------------------- K2: flash attention -> y
// Block = 32 query rows. Two passes over s: (1) rowmax, (2) exp/rowsum + y accumulation.
__global__ __launch_bounds__(256) void attn_kernel(
    const __bf16* __restrict__ Qt, const __bf16* __restrict__ Kt, const __bf16* __restrict__ Vc,
    float* __restrict__ y_f32, __bf16* __restrict__ y_bf) {
    constexpr int TT = 32, SB = 64;
    constexpr int LDS_S = SB + 4;   // f32 stride
    constexpr int LDP   = SB + 8;   // bf16 stride (144 B rows, 16B-aligned)
    __shared__ float  S_lds[TT * LDS_S];
    __shared__ __bf16 P_lds[TT * LDP];
    __shared__ float  rowstat[TT];
    __shared__ float  part[256];
    const int tid = threadIdx.x, lane = tid & 31, w = tid >> 5;
    const int b = blockIdx.y, t0 = blockIdx.x * TT;
    const __bf16* Qb = Qt + (size_t)b * T * C;
    const __bf16* Kb = Kt + (size_t)b * T * C;
    const __bf16* Vb = Vc + (size_t)b * C * T;
    const int msubS = w >> 2, nsubS = w & 3;          // score tile ownership (2m x 4n)
    const int srow = tid >> 3, schunk = (tid & 7) * 8; // softmax ownership: 8 threads/row
    const int mh = (lane >> 4) << 3, n = lane & 15;

    // ---- pass 1: row maxima of scaled scores
    float rm = -1e30f;
    for (int s0 = 0; s0 < T; s0 += SB) {
        v8f acc = {0, 0, 0, 0, 0, 0, 0, 0};
        const __bf16* Arow = Qb + (size_t)(t0 + msubS * 16) * C;
        const __bf16* Brow = Kb + (size_t)(s0 + nsubS * 16) * C;
        for (int k = 0; k < C; k += 32) {
            v16bf a  = load_a_frag(Arow + k, C, lane);
            v16bf bb = load_b_frag(Brow + k, C, lane);
            acc = wmma_bf16(a, bb, acc);
        }
        __syncthreads();
        for (int i = 0; i < 8; ++i)
            S_lds[(msubS * 16 + i + mh) * LDS_S + nsubS * 16 + n] = acc[i];
        __syncthreads();
        for (int j = 0; j < 8; ++j)
            rm = fmaxf(rm, S_lds[srow * LDS_S + schunk + j] * SCALE);
    }
    part[tid] = rm; __syncthreads();
    if ((tid & 7) == 0) {
        float m = part[tid];
        for (int j = 1; j < 8; ++j) m = fmaxf(m, part[tid + j]);
        rowstat[srow] = m;
    }
    __syncthreads();
    const float rowmax = rowstat[srow];
    __syncthreads();

    // ---- pass 2: P = exp(S - max), rowsum, y += P @ V
    v8f yacc[8];
    const v8f vzero = {0, 0, 0, 0, 0, 0, 0, 0};
    for (int j = 0; j < 8; ++j) yacc[j] = vzero;
    float rs = 0.f;
    for (int s0 = 0; s0 < T; s0 += SB) {
        v8f acc = vzero;
        const __bf16* Arow = Qb + (size_t)(t0 + msubS * 16) * C;
        const __bf16* Brow = Kb + (size_t)(s0 + nsubS * 16) * C;
        for (int k = 0; k < C; k += 32) {
            v16bf a  = load_a_frag(Arow + k, C, lane);
            v16bf bb = load_b_frag(Brow + k, C, lane);
            acc = wmma_bf16(a, bb, acc);
        }
        __syncthreads();
        for (int i = 0; i < 8; ++i)
            S_lds[(msubS * 16 + i + mh) * LDS_S + nsubS * 16 + n] = acc[i];
        __syncthreads();
        for (int j = 0; j < 8; ++j) {
            float p = __expf(S_lds[srow * LDS_S + schunk + j] * SCALE - rowmax);
            rs += p;
            P_lds[srow * LDP + schunk + j] = (__bf16)p;
        }
        __syncthreads();
        for (int j = 0; j < 8; ++j) {                  // wave: 2m x (4 strided n-tiles x 8)
            int c0 = ((w & 3) + 4 * j) * 16;
            for (int kk = 0; kk < 2; ++kk) {
                v16bf a  = load_a_frag(P_lds + (w >> 2) * 16 * LDP + kk * 32, LDP, lane);
                v16bf bb = load_b_frag(Vb + (size_t)c0 * T + s0 + kk * 32, T, lane);
                yacc[j] = wmma_bf16(a, bb, yacc[j]);
            }
        }
        __syncthreads();
    }
    part[tid] = rs; __syncthreads();
    if ((tid & 7) == 0) {
        float s = 0.f;
        for (int j = 0; j < 8; ++j) s += part[tid + j];
        rowstat[srow] = s;
    }
    __syncthreads();
    // scale by 1/rowsum and store y (f32 for scores, bf16 as next GEMM operand)
    for (int j = 0; j < 8; ++j) {
        int c0 = ((w & 3) + 4 * j) * 16;
        for (int i = 0; i < 8; ++i) {
            int m = (w >> 2) * 16 + i + mh;
            float val = yacc[j][i] / rowstat[m];
            size_t idx = ((size_t)b * T + t0 + m) * C + c0 + n;
            y_f32[idx] = val;
            y_bf[idx]  = (__bf16)val;
        }
    }
}

// ---------------------------------------------------------------- K3: out_raw = feat + Ws@y^T + bs (+ LN partials)
__global__ __launch_bounds__(256) void smooth_kernel(
    const float* __restrict__ feat, const __bf16* __restrict__ Ws, const float* __restrict__ bs,
    const __bf16* __restrict__ y_bf, float* __restrict__ out,
    float* __restrict__ part_sum, float* __restrict__ part_sumsq) {
    __shared__ float red1[256], red2[256];
    const int tid = threadIdx.x, lane = tid & 31, w = tid >> 5;
    const int b = blockIdx.z, c0 = blockIdx.y * 64, t0 = blockIdx.x * 64;
    const int mh = (lane >> 4) << 3, n = lane & 15;
    float ls = 0.f, lss = 0.f;
    for (int tile = w; tile < 16; tile += 8) {
        int osub = tile >> 2, tsub = tile & 3;
        v8f acc = {0, 0, 0, 0, 0, 0, 0, 0};
        const __bf16* Arow = Ws + (size_t)(c0 + osub * 16) * C;
        const __bf16* Brow = y_bf + ((size_t)b * T + t0 + tsub * 16) * C;
        for (int k = 0; k < C; k += 32) {
            v16bf a  = load_a_frag(Arow + k, C, lane);
            v16bf bb = load_b_frag(Brow + k, C, lane);
            acc = wmma_bf16(a, bb, acc);
        }
        for (int i = 0; i < 8; ++i) {
            int o = c0 + osub * 16 + i + mh;
            int t = t0 + tsub * 16 + n;
            size_t idx = ((size_t)b * C + o) * T + t;
            float val = acc[i] + bs[o] + feat[idx];
            out[idx] = val;
            ls += val; lss += val * val;
        }
    }
    red1[tid] = ls; red2[tid] = lss; __syncthreads();
    for (int s = 128; s > 0; s >>= 1) {
        if (tid < s) { red1[tid] += red1[tid + s]; red2[tid] += red2[tid + s]; }
        __syncthreads();
    }
    if (tid == 0) {
        int blk = (b * gridDim.y + blockIdx.y) * gridDim.x + blockIdx.x;
        part_sum[blk] = red1[0]; part_sumsq[blk] = red2[0];
    }
}

// ---------------------------------------------------------------- K5: per-batch LN stats + cb
__global__ __launch_bounds__(256) void stats_kernel(
    const float* __restrict__ part_sum, const float* __restrict__ part_sumsq,
    const float* __restrict__ vdot_part,
    float* __restrict__ ln_mu, float* __restrict__ ln_rs, float* __restrict__ cb) {
    __shared__ float r1[256], r2[256], r3[256];
    const int b = blockIdx.x, tid = threadIdx.x;
    r1[tid] = part_sum[b * 256 + tid];
    r2[tid] = part_sumsq[b * 256 + tid];
    r3[tid] = (tid < T / 32) ? vdot_part[b * (T / 32) + tid] : 0.f;
    __syncthreads();
    for (int s = 128; s > 0; s >>= 1) {
        if (tid < s) { r1[tid] += r1[tid + s]; r2[tid] += r2[tid + s]; r3[tid] += r3[tid + s]; }
        __syncthreads();
    }
    if (tid == 0) {
        float N = (float)C * (float)T;
        float mu = r1[0] / N;
        float var = r2[0] / N - mu * mu;
        ln_mu[b] = mu;
        ln_rs[b] = rsqrtf(var + LN_EPS);
        cb[b] = r3[0];
    }
}

// ---------------------------------------------------------------- K4: f_score
__global__ __launch_bounds__(256) void score_kernel(
    const float* __restrict__ y_f32, const float* __restrict__ Wl, const float* __restrict__ bl,
    const float* __restrict__ cb, float* __restrict__ fscore) {
    const int tid = threadIdx.x, lane = tid & 31, w = tid >> 5;
    const int b = blockIdx.y, t = blockIdx.x * 8 + w;
    const float* yr = y_f32 + ((size_t)b * T + t) * C;
    float u = 0.f;
    for (int c = lane; c < C; c += 32) u += yr[c] * Wl[c];
    for (int off = 16; off; off >>= 1) u += __shfl_xor(u, off, 32);
    if (lane == 0) {
        float bl0 = bl[0];
        float fs  = 1.f / (1.f + __expf(-(u + bl0)));
        float bsc = 1.f - 1.f / (1.f + __expf(-(cb[b] - u + bl0)));
        fscore[(size_t)b * T + t] = 0.5f * (fs + bsc);
    }
}

// ---------------------------------------------------------------- K6: LayerNorm apply
__global__ void lnapply_kernel(float* __restrict__ out,
                               const float* __restrict__ ln_mu, const float* __restrict__ ln_rs) {
    size_t i = (size_t)blockIdx.x * blockDim.x + threadIdx.x;
    if (i < (size_t)B * C * T) {
        int b = (int)(i / ((size_t)C * T));
        out[i] = (out[i] - ln_mu[b]) * ln_rs[b];
    }
}

// ----------------------------------------------------------------
extern "C" void kernel_launch(void* const* d_in, const int* in_sizes, int n_in,
                              void* d_out, int out_size, void* d_ws, size_t ws_size,
                              hipStream_t stream) {
    (void)in_sizes; (void)n_in; (void)out_size; (void)ws_size;
    const float* feat = (const float*)d_in[0];
    const float* Wq   = (const float*)d_in[1];
    const float* bq   = (const float*)d_in[2];
    const float* Wk   = (const float*)d_in[3];
    const float* bk   = (const float*)d_in[4];
    const float* Wv   = (const float*)d_in[5];
    const float* bv   = (const float*)d_in[6];
    const float* Ws   = (const float*)d_in[7];
    const float* bs   = (const float*)d_in[8];
    const float* Wl   = (const float*)d_in[9];
    const float* bl   = (const float*)d_in[10];
    float* out    = (float*)d_out;
    float* fscore = out + (size_t)B * C * T;

    char* ws = (char*)d_ws;
    size_t off = 0;
    auto alloc = [&](size_t bytes) -> void* {
        void* p = ws + off;
        off += (bytes + 255) & ~(size_t)255;
        return p;
    };
    __bf16* Wq_bf = (__bf16*)alloc((size_t)C * C * 2);
    __bf16* Wk_bf = (__bf16*)alloc((size_t)C * C * 2);
    __bf16* Wv_bf = (__bf16*)alloc((size_t)C * C * 2);
    __bf16* Ws_bf = (__bf16*)alloc((size_t)C * C * 2);
    __bf16* Qt    = (__bf16*)alloc((size_t)B * T * C * 2);
    __bf16* Kt    = (__bf16*)alloc((size_t)B * T * C * 2);
    __bf16* Vc    = (__bf16*)alloc((size_t)B * T * C * 2);
    float* y_f32  = (float*)alloc((size_t)B * T * C * 4);
    __bf16* y_bf  = (__bf16*)alloc((size_t)B * T * C * 2);
    float* vdot_part  = (float*)alloc((size_t)B * (T / 32) * 4);
    float* part_sum   = (float*)alloc((size_t)B * 256 * 4);
    float* part_sumsq = (float*)alloc((size_t)B * 256 * 4);
    float* ln_mu = (float*)alloc(B * 4);
    float* ln_rs = (float*)alloc(B * 4);
    float* cb    = (float*)alloc(B * 4);

    convert_weights<<<(C * C + 255) / 256, 256, 0, stream>>>(Wq, Wk, Wv, Ws, Wq_bf, Wk_bf, Wv_bf, Ws_bf);
    qkv_kernel<<<dim3(T / 32, B), 256, 0, stream>>>(feat, Wq_bf, Wk_bf, Wv_bf, bq, bk, bv, Wl,
                                                    Qt, Kt, Vc, vdot_part);
    attn_kernel<<<dim3(T / 32, B), 256, 0, stream>>>(Qt, Kt, Vc, y_f32, y_bf);
    smooth_kernel<<<dim3(T / 64, C / 64, B), 256, 0, stream>>>(feat, Ws_bf, bs, y_bf, out,
                                                               part_sum, part_sumsq);
    stats_kernel<<<B, 256, 0, stream>>>(part_sum, part_sumsq, vdot_part, ln_mu, ln_rs, cb);
    score_kernel<<<dim3(T / 8, B), 256, 0, stream>>>(y_f32, Wl, bl, cb, fscore);
    lnapply_kernel<<<(B * C * T) / 256, 256, 0, stream>>>(out, ln_mu, ln_rs);
}
